// SequentialGaussianBNN_70265664962762
// MI455X (gfx1250) — matrix operand order repaired and checked by reading, not compiled, not run on hardware
//
#include <hip/hip_runtime.h>
#include <hip/hip_bf16.h>
#include <cstdint>
#include <cstddef>

// ---------------------------------------------------------------------------
// CDNA5 (gfx1250) WMMA types
// ---------------------------------------------------------------------------
typedef __attribute__((ext_vector_type(16))) __bf16 v16bf;
typedef __attribute__((ext_vector_type(8)))  float  v8f;
typedef int v4i __attribute__((ext_vector_type(4)));

#define BM 128
#define BN 128
#define BK 32
#define NTHREADS 256

#if defined(__gfx1250__) && __has_builtin(__builtin_amdgcn_global_load_async_to_lds_b128)
#define USE_ASYNC_LDS 1
#else
#define USE_ASYNC_LDS 0
#endif

// Builtin signature (from hipcc diagnostic): (v4i as1*, v4i as3*, imm offset, imm cpol)
typedef __attribute__((address_space(1))) v4i* as1_v4i_p;
typedef __attribute__((address_space(3))) v4i* as3_v4i_p;
#define AS1(p) ((as1_v4i_p)(v4i*)(void*)(p))
#define AS3(p) ((as3_v4i_p)(v4i*)(void*)(p))

__device__ __forceinline__ float softplus_f(float x) {
    // numerically stable: max(x,0) + log1p(exp(-|x|))
    return fmaxf(x, 0.0f) + log1pf(expf(-fabsf(x)));
}

// ---------------------------------------------------------------------------
// Elementwise kernels (negligible cost vs the GEMMs)
// ---------------------------------------------------------------------------
__global__ void materialize_weights(const float* __restrict__ mw,
                                    const float* __restrict__ sw,
                                    const float* __restrict__ zw,
                                    __bf16* __restrict__ out, int n) {
    int i = blockIdx.x * blockDim.x + threadIdx.x;
    if (i < n) out[i] = (__bf16)(mw[i] + softplus_f(sw[i]) * zw[i]);
}

__global__ void materialize_bias(const float* __restrict__ mb,
                                 const float* __restrict__ sb,
                                 const float* __restrict__ zb,
                                 float* __restrict__ out, int n) {
    int i = blockIdx.x * blockDim.x + threadIdx.x;
    if (i < n) out[i] = mb[i] + softplus_f(sb[i]) * zb[i];
}

__global__ void cast_to_bf16(const float* __restrict__ in,
                             __bf16* __restrict__ out, int n) {
    int i = blockIdx.x * blockDim.x + threadIdx.x;
    if (i < n) out[i] = (__bf16)in[i];
}

// ---------------------------------------------------------------------------
// Fused GEMM:  out[M,N] = H[M,K] * W[N,K]^T + bias, optional tanh
//   H, W in bf16 row-major; acc in f32 via v_wmma_f32_16x16x32_bf16.
//   256 threads = 8 waves; block tile 128x128; wave tile 64x32 (4x2 WMMA).
// ---------------------------------------------------------------------------
union FragB { v16bf v; uint32_t u[8]; };

template <bool LAST>
__global__ __launch_bounds__(NTHREADS)
void gemm_bnn(const __bf16* __restrict__ H,
              const __bf16* __restrict__ W,
              const float*  __restrict__ bias,
              void* __restrict__ outp,
              int M, int N, int K)
{
    __shared__ __align__(16) __bf16 sH[2][BM][BK];
    __shared__ __align__(16) __bf16 sW[2][BN][BK];

    const int tid  = threadIdx.x;
    const int lane = tid & 31;
    const int wave = tid >> 5;
    const int m_wave = (wave >> 2) * 64;   // 0 or 64
    const int n_wave = (wave & 3) * 32;    // 0,32,64,96

    const int blockN = blockIdx.x * BN;
    const int blockM = blockIdx.y * BM;

    // staging: each thread moves 32B of H-tile and 32B of W-tile per K-step
    const int sr = tid >> 1;            // tile row 0..127
    const int sc = (tid & 1) * 16;      // bf16 col offset: 0 or 16

    const __bf16* gH = H + (size_t)(blockM + sr) * K + sc;
    const __bf16* gW = W + (size_t)(blockN + sr) * K + sc;

    v8f acc[4][2];
    {
        v8f zero = {0.f, 0.f, 0.f, 0.f, 0.f, 0.f, 0.f, 0.f};
#pragma unroll
        for (int i = 0; i < 4; ++i)
#pragma unroll
            for (int j = 0; j < 2; ++j) acc[i][j] = zero;
    }

    auto stage = [&](int kt, int bi) {
        const __bf16* ph = gH + (size_t)kt * BK;
        const __bf16* pw = gW + (size_t)kt * BK;
        __bf16* lh = &sH[bi][sr][sc];
        __bf16* lw = &sW[bi][sr][sc];
#if USE_ASYNC_LDS
        __builtin_amdgcn_global_load_async_to_lds_b128(AS1(ph),     AS3(lh),     0, 0);
        __builtin_amdgcn_global_load_async_to_lds_b128(AS1(ph + 8), AS3(lh + 8), 0, 0);
        __builtin_amdgcn_global_load_async_to_lds_b128(AS1(pw),     AS3(lw),     0, 0);
        __builtin_amdgcn_global_load_async_to_lds_b128(AS1(pw + 8), AS3(lw + 8), 0, 0);
#else
        uint4 a0 = *(const uint4*)ph;
        uint4 a1 = *(const uint4*)(ph + 8);
        uint4 b0 = *(const uint4*)pw;
        uint4 b1 = *(const uint4*)(pw + 8);
        *(uint4*)lh       = a0;
        *(uint4*)(lh + 8) = a1;
        *(uint4*)lw       = b0;
        *(uint4*)(lw + 8) = b1;
#endif
    };

    auto wait_stage = [&]() {
#if USE_ASYNC_LDS
#if __has_builtin(__builtin_amdgcn_s_wait_asynccnt)
        __builtin_amdgcn_s_wait_asynccnt(0);
#else
        asm volatile("s_wait_asynccnt 0" ::: "memory");
#endif
#endif
        __syncthreads();
    };

    const int ksteps = K / BK;
    stage(0, 0);
    wait_stage();

    const int am = lane & 15;   // M (or N) index within 16x16 tile
    const int kg = lane >> 4;   // K half selector

    for (int kt = 0; kt < ksteps; ++kt) {
        const int bi = kt & 1;
        if (kt + 1 < ksteps) stage(kt + 1, bi ^ 1);

        FragB a[4], b[2];
        // A-frag (16x32 bf16): lanes 0-15 -> K 0..7 & 16..23; lanes 16-31 -> K 8..15 & 24..31
#pragma unroll
        for (int i = 0; i < 4; ++i) {
            const __bf16* p = &sH[bi][m_wave + i * 16 + am][kg * 8];
            *(uint4*)&a[i].u[0] = *(const uint4*)p;          // VGPRs 0-3
            *(uint4*)&a[i].u[4] = *(const uint4*)(p + 16);   // VGPRs 4-7
        }
        // B-frag (32x16 bf16): lanes 0-15 -> K 0..7 & 16..23; lanes 16-31 -> K 8..15 & 24..31
#pragma unroll
        for (int j = 0; j < 2; ++j) {
            const __bf16* p = &sW[bi][n_wave + j * 16 + am][kg * 8];
            *(uint4*)&b[j].u[0] = *(const uint4*)p;
            *(uint4*)&b[j].u[4] = *(const uint4*)(p + 16);
        }

#pragma unroll
        for (int i = 0; i < 4; ++i)
#pragma unroll
            for (int j = 0; j < 2; ++j)
                acc[i][j] = __builtin_amdgcn_wmma_f32_16x16x32_bf16(
                    /*neg_a=*/false, a[i].v, /*neg_b=*/false, b[j].v,
                    /*c_mod=*/(short)0, acc[i][j],
                    /*reuse_a=*/false, /*reuse_b=*/false);

        wait_stage();
    }

    // Epilogue: C/D layout — VGPR v: lanes0-15 M=v, lanes16-31 M=v+8; N=lane%16
    const int cn = lane & 15;
    const int mh = lane >> 4;
#pragma unroll
    for (int i = 0; i < 4; ++i) {
#pragma unroll
        for (int j = 0; j < 2; ++j) {
            const int col = blockN + n_wave + j * 16 + cn;
            const float bv = bias[col];
#pragma unroll
            for (int v = 0; v < 8; ++v) {
                const int row = blockM + m_wave + i * 16 + mh * 8 + v;
                float r = acc[i][j][v] + bv;
                if (!LAST) {
                    r = tanhf(r);
                    ((__bf16*)outp)[(size_t)row * N + col] = (__bf16)r;
                } else {
                    ((float*)outp)[(size_t)row * N + col] = r;
                }
            }
        }
    }
}

// ---------------------------------------------------------------------------
// Launch
// ---------------------------------------------------------------------------
extern "C" void kernel_launch(void* const* d_in, const int* in_sizes, int n_in,
                              void* d_out, int out_size, void* d_ws, size_t ws_size,
                              hipStream_t stream)
{
    const int B = 8192;
    const int S[5] = {1024, 4096, 4096, 4096, 1024};

    char* ws = (char*)d_ws;
    size_t off = 0;
    auto wsalloc = [&](size_t bytes) -> void* {
        void* p = ws + off;
        off += (bytes + 255) & ~(size_t)255;
        return p;
    };

    __bf16* act0 = (__bf16*)wsalloc((size_t)B * 1024 * 2);
    __bf16* actA = (__bf16*)wsalloc((size_t)B * 4096 * 2);
    __bf16* actB = (__bf16*)wsalloc((size_t)B * 4096 * 2);
    __bf16* Wd[4];
    float*  bd[4];
    for (int j = 0; j < 4; ++j) {
        Wd[j] = (__bf16*)wsalloc((size_t)S[j + 1] * S[j] * 2);
        bd[j] = (float*)wsalloc((size_t)S[j + 1] * 4);
    }

    // Materialize reparameterized weights/biases (bf16 weights, f32 biases)
    for (int j = 0; j < 4; ++j) {
        const float* mw = (const float*)d_in[1 + 6 * j + 0];
        const float* sw = (const float*)d_in[1 + 6 * j + 1];
        const float* zw = (const float*)d_in[1 + 6 * j + 2];
        const float* mb = (const float*)d_in[1 + 6 * j + 3];
        const float* sb = (const float*)d_in[1 + 6 * j + 4];
        const float* zb = (const float*)d_in[1 + 6 * j + 5];
        const int nW = S[j + 1] * S[j];
        materialize_weights<<<(nW + 255) / 256, 256, 0, stream>>>(mw, sw, zw, Wd[j], nW);
        materialize_bias<<<(S[j + 1] + 255) / 256, 256, 0, stream>>>(mb, sb, zb, bd[j], S[j + 1]);
    }

    const float* x = (const float*)d_in[0];
    cast_to_bf16<<<(B * 1024 + 255) / 256, 256, 0, stream>>>(x, act0, B * 1024);

    dim3 blk(NTHREADS);
    // layer 0: [8192,1024] x [4096,1024]^T -> actA (tanh)
    gemm_bnn<false><<<dim3(4096 / BN, B / BM), blk, 0, stream>>>(act0, Wd[0], bd[0], actA, B, 4096, 1024);
    // layer 1
    gemm_bnn<false><<<dim3(4096 / BN, B / BM), blk, 0, stream>>>(actA, Wd[1], bd[1], actB, B, 4096, 4096);
    // layer 2
    gemm_bnn<false><<<dim3(4096 / BN, B / BM), blk, 0, stream>>>(actB, Wd[2], bd[2], actA, B, 4096, 4096);
    // layer 3: no tanh, fp32 out
    gemm_bnn<true><<<dim3(1024 / BN, B / BM), blk, 0, stream>>>(actA, Wd[3], bd[3], d_out, B, 1024, 4096);
}
